// Network_56427280335153
// MI455X (gfx1250) — matrix operand order repaired and checked by reading, not compiled, not run on hardware
//
#include <hip/hip_runtime.h>

// ---------------------------------------------------------------------------
// Types for CDNA5 WMMA (wave32, v_wmma_f32_16x16x32_bf16)
// ---------------------------------------------------------------------------
typedef __bf16 bf16_t;
typedef bf16_t v8bf  __attribute__((ext_vector_type(8)));
typedef bf16_t v16bf __attribute__((ext_vector_type(16)));
typedef float  v8f   __attribute__((ext_vector_type(8)));
typedef int    v4i_t __attribute__((ext_vector_type(4)));

#if __has_builtin(__builtin_amdgcn_global_load_async_to_lds_b128) && \
    __has_builtin(__builtin_amdgcn_s_wait_asynccnt)
#define USE_ASYNC 1
#else
#define USE_ASYNC 0
#endif

#define NB      16
#define NCTX    2048
#define NLAT    512
#define NQ      4096
#define DDIM    512
#define NHEAD   8
#define DHEAD   64
#define NDEPTH  4

// ---------------------------------------------------------------------------
// Generic batched GEMM:  C[z] = A[z](MxK, f32) * Bt[z](NxK, bf16)^T  (+ epi)
// Block: 256 thr (8 waves), tile 128x64, BK=64, wave tile 32x32 (2x2 WMMA,
// 2 K-substeps => 8 WMMA per wave per barrier), double-buffered LDS.
// B operand staged via GLOBAL_LOAD_ASYNC_TO_LDS (ASYNCcnt) when available.
// ---------------------------------------------------------------------------
struct GP {
  const float*  A;        // f32 activations
  const bf16_t* Bt;       // bf16, row n holds K contiguous elements
  float*        C;
  const float*  bias;     // [N] or null
  const float*  resid;    // same layout as C, or null
  const float*  rowScale; // per-row scalar (PointEmbed values) or null
  const float*  vew;      // [N]
  const float*  veb;      // [N]
  bf16_t*       ksp;      // EPI=1: k output  [M][512] bf16
  bf16_t*       vT;       // EPI=1: v^T output [(b*H+h)*dh+d][512] bf16
  long aO, aI, bO, bI, cO, cI;
  int  M, N, K, lda, ldbt, ldc, zdiv, H, dh;
};

#define LDA_S 72
#define LDB_S 72

template <int EPI>
__global__ __launch_bounds__(256) void gemm_k(GP p) {
  __shared__ bf16_t lA[2][128 * LDA_S];
  __shared__ bf16_t lB[2][64 * LDB_S];

  const int tid = threadIdx.x;
  const int z   = blockIdx.z;
  const int zo  = z / p.zdiv, zi = z - zo * p.zdiv;
  const long abase = (long)zo * p.aO + (long)zi * p.aI;
  const long bbase = (long)zo * p.bO + (long)zi * p.bI;
  const long cbase = (long)zo * p.cO + (long)zi * p.cI;
  const int m0 = blockIdx.y * 128;
  const int n0 = blockIdx.x * 64;

  const int lane = tid & 31, wv = tid >> 5;
  const int wm = wv >> 1, wn = wv & 1;       // 4x2 wave grid
  const int lh = lane >> 4, l15 = lane & 15;

  v8f acc[2][2];
#pragma unroll
  for (int i = 0; i < 2; i++)
#pragma unroll
    for (int j = 0; j < 2; j++)
#pragma unroll
      for (int r = 0; r < 8; r++) acc[i][j][r] = 0.f;

  const int ar = tid >> 1, ak = (tid & 1) * 32;   // A stage: 128 rows x 64
  const int br = tid >> 2, bk = (tid & 3) * 16;   // B stage: 64 rows x 64
  const float*  Ab = p.A + abase;
  const bf16_t* Bb = p.Bt + bbase;

  float4 fa[8];

#define LOAD_A(K0)                                                            \
  {                                                                           \
    const float* ga = Ab + (long)(m0 + ar) * p.lda + (K0) + ak;               \
    _Pragma("unroll") for (int q = 0; q < 8; q++)                             \
        fa[q] = *(const float4*)(ga + q * 4);                                 \
  }

#define STORE_A(BUF)                                                          \
  {                                                                           \
    _Pragma("unroll") for (int q = 0; q < 4; q++) {                           \
      v8bf h;                                                                 \
      h[0] = (bf16_t)fa[2 * q].x;     h[1] = (bf16_t)fa[2 * q].y;             \
      h[2] = (bf16_t)fa[2 * q].z;     h[3] = (bf16_t)fa[2 * q].w;             \
      h[4] = (bf16_t)fa[2 * q + 1].x; h[5] = (bf16_t)fa[2 * q + 1].y;         \
      h[6] = (bf16_t)fa[2 * q + 1].z; h[7] = (bf16_t)fa[2 * q + 1].w;         \
      *(v8bf*)&lA[BUF][ar * LDA_S + ak + q * 8] = h;                          \
    }                                                                         \
  }

#if USE_ASYNC
#define STAGE_B(K0, BUF)                                                      \
  {                                                                           \
    const bf16_t* gp = Bb + (long)(n0 + br) * p.ldbt + (K0) + bk;             \
    __builtin_amdgcn_global_load_async_to_lds_b128(                           \
        (v4i_t*)(void*)gp,                                                    \
        (v4i_t*)(void*)&lB[BUF][br * LDB_S + bk], 0, 0);                      \
    __builtin_amdgcn_global_load_async_to_lds_b128(                           \
        (v4i_t*)(void*)(gp + 8),                                              \
        (v4i_t*)(void*)&lB[BUF][br * LDB_S + bk + 8], 0, 0);                  \
  }
#define WAIT_B() __builtin_amdgcn_s_wait_asynccnt(0)
#else
#define STAGE_B(K0, BUF)                                                      \
  {                                                                           \
    const bf16_t* gp = Bb + (long)(n0 + br) * p.ldbt + (K0) + bk;             \
    *(v8bf*)&lB[BUF][br * LDB_S + bk]     = *(const v8bf*)(gp);               \
    *(v8bf*)&lB[BUF][br * LDB_S + bk + 8] = *(const v8bf*)(gp + 8);           \
  }
#define WAIT_B()
#endif

#define COMPUTE(BUF)                                                          \
  {                                                                           \
    const bf16_t* La = &lA[BUF][0];                                           \
    const bf16_t* Lb = &lB[BUF][0];                                           \
    _Pragma("unroll") for (int s = 0; s < 2; s++) {                           \
      v16bf af[2], bfv[2];                                                    \
      _Pragma("unroll") for (int i = 0; i < 2; i++) {                         \
        int row = wm * 32 + i * 16 + l15;                                     \
        v8bf lo = *(const v8bf*)&La[row * LDA_S + s * 32 + lh * 8];           \
        v8bf hi = *(const v8bf*)&La[row * LDA_S + s * 32 + 16 + lh * 8];      \
        af[i] = __builtin_shufflevector(lo, hi, 0, 1, 2, 3, 4, 5, 6, 7, 8, 9, \
                                        10, 11, 12, 13, 14, 15);              \
      }                                                                       \
      _Pragma("unroll") for (int j = 0; j < 2; j++) {                         \
        int row = wn * 32 + j * 16 + l15;                                     \
        v8bf lo = *(const v8bf*)&Lb[row * LDB_S + s * 32 + lh * 8];           \
        v8bf hi = *(const v8bf*)&Lb[row * LDB_S + s * 32 + 16 + lh * 8];      \
        bfv[j] = __builtin_shufflevector(lo, hi, 0, 1, 2, 3, 4, 5, 6, 7, 8,   \
                                         9, 10, 11, 12, 13, 14, 15);          \
      }                                                                       \
      _Pragma("unroll") for (int i = 0; i < 2; i++)                           \
          _Pragma("unroll") for (int j = 0; j < 2; j++)                       \
              acc[i][j] = __builtin_amdgcn_wmma_f32_16x16x32_bf16(            \
                  false, af[i], false, bfv[j], (short)0, acc[i][j], false,    \
                  false);                                                     \
    }                                                                         \
  }

  // ---- prologue: stage tile 0 ----
  LOAD_A(0);
  STAGE_B(0, 0);
  STORE_A(0);
  WAIT_B();
  __syncthreads();

  int cur = 0;
  const int nIter = p.K >> 6;
  for (int it = 1; it < nIter; it++) {
    const int k0 = it * 64;
    LOAD_A(k0);                                   // next tile -> regs (global)
    __builtin_prefetch(Ab + (long)(m0 + ar) * p.lda + k0 + 64 + ak, 0, 3);
    STAGE_B(k0, cur ^ 1);                         // next tile -> LDS (async)
    COMPUTE(cur);                                 // 8 WMMA on current tile
    STORE_A(cur ^ 1);
    WAIT_B();
    __syncthreads();
    cur ^= 1;
  }
  COMPUTE(cur);

  // ---- epilogue (D layout: VGPR r -> M = r + lh*8, lane&15 -> N) ----
#pragma unroll
  for (int i = 0; i < 2; i++)
#pragma unroll
    for (int j = 0; j < 2; j++)
#pragma unroll
      for (int r = 0; r < 8; r++) {
        int m = m0 + wm * 32 + i * 16 + lh * 8 + r;
        int n = n0 + wn * 32 + j * 16 + l15;
        float v = acc[i][j][r];
        if (EPI == 0) {
          if (p.bias) v += p.bias[n];
          if (p.rowScale) v += p.rowScale[m] * p.vew[n] + p.veb[n];
          long idx = cbase + (long)m * p.ldc + n;
          if (p.resid) v += p.resid[idx];
          p.C[idx] = v;
        } else {  // kv split: n<512 -> k bf16 ; n>=512 -> v^T bf16
          if (n < 512) {
            p.ksp[(long)m * 512 + n] = (bf16_t)v;
          } else {
            int b = m >> 9, jj = m & 511;
            int nn = n - 512;
            int hh = nn / p.dh, d = nn - hh * p.dh;
            p.vT[(((long)b * p.H + hh) * p.dh + d) * 512 + jj] = (bf16_t)v;
          }
        }
      }
#undef LOAD_A
#undef STORE_A
#undef STAGE_B
#undef WAIT_B
#undef COMPUTE
}

// ---------------------------------------------------------------------------
// Small helper kernels
// ---------------------------------------------------------------------------
__global__ void convT_k(const float* __restrict__ W, bf16_t* __restrict__ WT,
                        int K, int N) {
  long idx = (long)blockIdx.x * 256 + threadIdx.x;
  if (idx >= (long)K * N) return;
  int n = (int)(idx / K);
  int k = (int)(idx - (long)n * K);
  WT[idx] = (bf16_t)W[(long)k * N + n];
}

__global__ void pe_convT_k(const float* __restrict__ W, bf16_t* __restrict__ WT) {
  int idx = blockIdx.x * 256 + threadIdx.x;  // 512*64
  int n = idx >> 6, k = idx & 63;
  WT[idx] = (k < 51) ? (bf16_t)W[k * 512 + n] : (bf16_t)0.f;
}

__global__ void label_k(const float* __restrict__ alpha,
                        const float* __restrict__ freqs, float* __restrict__ label) {
  int idx = blockIdx.x * 256 + threadIdx.x;  // 16*512
  int b = idx >> 9, jj = idx & 511;
  float a = alpha[b] * 6.283185307179586f;
  label[idx] = (jj < 256) ? cosf(a * freqs[jj]) : sinf(a * freqs[jj - 256]);
}

__global__ void film_k(const float* __restrict__ label, const float* __restrict__ G,
                       float* __restrict__ out) {
  int idx = blockIdx.x * 256 + threadIdx.x;  // 16*512
  int b = idx >> 9, n = idx & 511;
  const float* l = label + b * 512;
  float s = 0.f;
  for (int k = 0; k < 512; k++) s += l[k] * G[k * 512 + n];
  out[idx] = s;
}

__global__ void embed_k(const float* __restrict__ pts, float* __restrict__ emb, int T) {
  int idx = blockIdx.x * 256 + threadIdx.x;
  if (idx >= T * 64) return;
  int t = idx >> 6, o = idx & 63;
  float v;
  if (o < 48) {
    int s = (o < 24) ? o : o - 24;
    int axis = s >> 3, jj = s & 7;
    float pr = pts[t * 3 + axis] * (3.14159265358979323846f * (float)(1 << jj));
    v = (o < 24) ? sinf(pr) : cosf(pr);
  } else if (o < 51) {
    v = pts[t * 3 + (o - 48)];
  } else {
    v = 0.f;
  }
  emb[(long)t * 64 + o] = v;
}

__global__ void bcast_k(const float* __restrict__ lat, float* __restrict__ x) {
  long idx = (long)blockIdx.x * 256 + threadIdx.x;  // 16*262144
  x[idx] = lat[idx & 262143];
}

__global__ void ln_k(const float* __restrict__ src, long srcB, int rowsPerB,
                     const float* __restrict__ s, const float* __restrict__ bsh,
                     const float* __restrict__ gamma, const float* __restrict__ beta,
                     float* __restrict__ dst) {
  __shared__ float red[256];
  int r = blockIdx.x;
  int b = r / rowsPerB;
  int j = r - b * rowsPerB;
  const float* x = src + (long)b * srcB + (long)j * 512;
  int t = threadIdx.x;
  float v0 = x[t], v1 = x[t + 256];
  red[t] = v0 + v1;
  __syncthreads();
  for (int st = 128; st > 0; st >>= 1) {
    if (t < st) red[t] += red[t + st];
    __syncthreads();
  }
  float mean = red[0] * (1.f / 512.f);
  __syncthreads();
  float d0 = v0 - mean, d1 = v1 - mean;
  red[t] = d0 * d0 + d1 * d1;
  __syncthreads();
  for (int st = 128; st > 0; st >>= 1) {
    if (t < st) red[t] += red[t + st];
    __syncthreads();
  }
  float rstd = rsqrtf(red[0] * (1.f / 512.f) + 1e-5f);
  float o0 = d0 * rstd * s[t] + bsh[t];
  float o1 = d1 * rstd * s[t + 256] + bsh[t + 256];
  if (gamma) {
    const float* g = gamma + (long)b * 512;
    const float* be = beta + (long)b * 512;
    o0 = g[t] * o0 + be[t];
    o1 = g[t + 256] * o1 + be[t + 256];
  }
  float* d = dst + (long)r * 512;
  d[t] = o0;
  d[t + 256] = o1;
}

__global__ void softmax_k(float* __restrict__ S, float scale) {
  __shared__ float red[256];
  float* p = S + (long)blockIdx.x * 512;
  int t = threadIdx.x;
  float v0 = p[t] * scale, v1 = p[t + 256] * scale;
  red[t] = fmaxf(v0, v1);
  __syncthreads();
  for (int st = 128; st > 0; st >>= 1) {
    if (t < st) red[t] = fmaxf(red[t], red[t + st]);
    __syncthreads();
  }
  float m = red[0];
  __syncthreads();
  float e0 = __expf(v0 - m), e1 = __expf(v1 - m);
  red[t] = e0 + e1;
  __syncthreads();
  for (int st = 128; st > 0; st >>= 1) {
    if (t < st) red[t] += red[t + st];
    __syncthreads();
  }
  float inv = 1.f / red[0];
  p[t] = e0 * inv;
  p[t + 256] = e1 * inv;
}

__global__ void geglu_k(const float* __restrict__ h, float* __restrict__ ag) {
  long idx = (long)blockIdx.x * 256 + threadIdx.x;  // Mtot*2048
  long m = idx >> 11;
  int c = idx & 2047;
  float a = h[m * 4096 + c];
  float g = h[m * 4096 + 2048 + c];
  float ge = 0.5f * g * (1.f + erff(g * 0.7071067811865475f));
  ag[idx] = a * ge;
}

__global__ void gemv_out_k(const float* __restrict__ on, const float* __restrict__ w,
                           const float* __restrict__ b, float* __restrict__ out) {
  __shared__ float red[128];
  long r = blockIdx.x;
  int t = threadIdx.x;
  const float* x = on + r * 512;
  float s = 0.f;
  for (int c = t; c < 512; c += 128) s += x[c] * w[c];
  red[t] = s;
  __syncthreads();
  for (int st = 64; st > 0; st >>= 1) {
    if (t < st) red[t] += red[t + st];
    __syncthreads();
  }
  if (t == 0) out[r] = red[0] + b[0];
}

// ---------------------------------------------------------------------------
// Input index map (setup_inputs() dict insertion order, params recursively)
// ---------------------------------------------------------------------------
enum {
  I_CPTS = 0, I_CVALS, I_QUERIES, I_ALPHA,
  I_FREQS, I_PE_W, I_PE_B, I_VE_W, I_VE_B, I_LATENT,
  I_CA_LN_S, I_CA_LNC_S, I_CF_LN_S, I_SA_LN_S, I_SF_LN_S,
  I_CA_LN_B, I_CA_LNC_B, I_CF_LN_B, I_SA_LN_B, I_SF_LN_B,
  I_CA_BO, I_SA_BO, I_CF_B2, I_SF_B2,
  I_CF_B1, I_SF_B1,
  I_CA_WQ, I_CA_WKV, I_CA_WO, I_CF_W1, I_CF_W2,
  I_SA_G, I_SA_BE, I_SA_WQ, I_SA_WKV, I_SA_WO,
  I_SF_G, I_SF_BE, I_SF_W1, I_SF_W2,
  I_DEC_LN_S, I_DEC_LN_B, I_DEC_LNC_S, I_DEC_LNC_B,
  I_DEC_WQ, I_DEC_WKV, I_DEC_WO, I_DEC_BO,
  I_OUT_LN_S, I_OUT_LN_B, I_OUT_G, I_OUT_BE, I_OUT_W, I_OUT_B
};

extern "C" void kernel_launch(void* const* d_in, const int* in_sizes, int n_in,
                              void* d_out, int out_size, void* d_ws, size_t ws_size,
                              hipStream_t stream) {
  (void)in_sizes; (void)out_size; (void)ws_size;
  const float* IN[64];
  for (int i = 0; i < 54 && i < n_in; i++) IN[i] = (const float*)d_in[i];

  // ---- workspace arena ----
  char* ws = (char*)d_ws;
  size_t off = 0;
  auto alloc = [&](size_t bytes) -> void* {
    void* p = ws + off;
    off = (off + bytes + 255) & ~(size_t)255;
    return p;
  };
  bf16_t *wCaWq[4], *wCaWkv[4], *wCaWo[4], *wCfW1[4], *wCfW2[4];
  bf16_t *wSaWq[4], *wSaWkv[4], *wSaWo[4], *wSfW1[4], *wSfW2[4];
  for (int i = 0; i < 4; i++) {
    wCaWq[i]  = (bf16_t*)alloc(262144L * 2);
    wCaWkv[i] = (bf16_t*)alloc(524288L * 2);
    wCaWo[i]  = (bf16_t*)alloc(262144L * 2);
    wCfW1[i]  = (bf16_t*)alloc(2097152L * 2);
    wCfW2[i]  = (bf16_t*)alloc(1048576L * 2);
    wSaWq[i]  = (bf16_t*)alloc(262144L * 2);
    wSaWkv[i] = (bf16_t*)alloc(524288L * 2);
    wSaWo[i]  = (bf16_t*)alloc(262144L * 2);
    wSfW1[i]  = (bf16_t*)alloc(2097152L * 2);
    wSfW2[i]  = (bf16_t*)alloc(1048576L * 2);
  }
  bf16_t* wDecWq  = (bf16_t*)alloc(262144L * 2);
  bf16_t* wDecWkv = (bf16_t*)alloc(524288L * 2);
  bf16_t* wDecWo  = (bf16_t*)alloc(262144L * 2);
  bf16_t* wPe     = (bf16_t*)alloc(32768L * 2);

  float* label = (float*)alloc(8192L * 4);
  float* film  = (float*)alloc(18L * 8192 * 4);
  float* emb   = (float*)alloc(65536L * 64 * 4);
  float* ctx   = (float*)alloc(16777216L * 4);    // [16][2048][512]
  float* xbuf  = (float*)alloc(4194304L * 4);     // [16][512][512]
  float* xn    = (float*)alloc(4194304L * 4);
  float* cnb   = (float*)alloc(4194304L * 4);
  float* qbuf  = (float*)alloc(33554432L * 4);    // up to [16][4096][512]
  float* Sbuf  = (float*)alloc(33554432L * 4);    // scores / GEGLU hidden / qe
  float* obuf  = (float*)alloc(33554432L * 4);    // attn out / GEGLU a*gelu(g)
  bf16_t* kbuf = (bf16_t*)alloc(4194304L * 2);
  bf16_t* vTb  = (bf16_t*)alloc(4194304L * 2);

  // ---- one-time weight conversion (f32 [K][N] -> bf16 [N][K]) ----
  auto convT = [&](const float* W, bf16_t* WT, int K, int N) {
    long tot = (long)K * N;
    convT_k<<<dim3((unsigned)((tot + 255) / 256)), 256, 0, stream>>>(W, WT, K, N);
  };
  for (int i = 0; i < 4; i++) {
    convT(IN[I_CA_WQ]  + (long)i * 262144,  wCaWq[i],  512, 512);
    convT(IN[I_CA_WKV] + (long)i * 524288,  wCaWkv[i], 512, 1024);
    convT(IN[I_CA_WO]  + (long)i * 262144,  wCaWo[i],  512, 512);
    convT(IN[I_CF_W1]  + (long)i * 2097152, wCfW1[i],  512, 4096);
    convT(IN[I_CF_W2]  + (long)i * 1048576, wCfW2[i],  2048, 512);
    convT(IN[I_SA_WQ]  + (long)i * 262144,  wSaWq[i],  512, 512);
    convT(IN[I_SA_WKV] + (long)i * 524288,  wSaWkv[i], 512, 1024);
    convT(IN[I_SA_WO]  + (long)i * 262144,  wSaWo[i],  512, 512);
    convT(IN[I_SF_W1]  + (long)i * 2097152, wSfW1[i],  512, 4096);
    convT(IN[I_SF_W2]  + (long)i * 1048576, wSfW2[i],  2048, 512);
  }
  convT(IN[I_DEC_WQ], wDecWq, 512, 512);
  convT(IN[I_DEC_WKV], wDecWkv, 512, 1024);
  convT(IN[I_DEC_WO], wDecWo, 512, 512);
  pe_convT_k<<<128, 256, 0, stream>>>(IN[I_PE_W], wPe);

  // ---- label + FiLM vectors ----
  label_k<<<32, 256, 0, stream>>>(IN[I_ALPHA], IN[I_FREQS], label);
  for (int i = 0; i < 4; i++) {
    film_k<<<32, 256, 0, stream>>>(label, IN[I_SA_G]  + (long)i * 262144, film + (i * 4 + 0) * 8192L);
    film_k<<<32, 256, 0, stream>>>(label, IN[I_SA_BE] + (long)i * 262144, film + (i * 4 + 1) * 8192L);
    film_k<<<32, 256, 0, stream>>>(label, IN[I_SF_G]  + (long)i * 262144, film + (i * 4 + 2) * 8192L);
    film_k<<<32, 256, 0, stream>>>(label, IN[I_SF_BE] + (long)i * 262144, film + (i * 4 + 3) * 8192L);
  }
  film_k<<<32, 256, 0, stream>>>(label, IN[I_OUT_G],  film + 16L * 8192);
  film_k<<<32, 256, 0, stream>>>(label, IN[I_OUT_BE], film + 17L * 8192);

  // ---- GEMM launchers ----
  auto gemm = [&](const float* A, int lda, long aO, long aI,
                  const bf16_t* Bt, int ldbt, long bO, long bI,
                  float* C, int ldc, long cO, long cI,
                  int M, int N, int K, int Z, int zdiv,
                  const float* bias, const float* resid,
                  const float* rowScale, const float* vw, const float* vb) {
    GP p{};
    p.A = A; p.lda = lda; p.aO = aO; p.aI = aI;
    p.Bt = Bt; p.ldbt = ldbt; p.bO = bO; p.bI = bI;
    p.C = C; p.ldc = ldc; p.cO = cO; p.cI = cI;
    p.M = M; p.N = N; p.K = K; p.zdiv = zdiv;
    p.bias = bias; p.resid = resid;
    p.rowScale = rowScale; p.vew = vw; p.veb = vb;
    dim3 g(N / 64, M / 128, Z);
    gemm_k<0><<<g, 256, 0, stream>>>(p);
  };
  auto gemmKV = [&](const float* A, const bf16_t* Bt, int M, int H, int dh) {
    GP p{};
    p.A = A; p.lda = 512; p.Bt = Bt; p.ldbt = 512;
    p.M = M; p.N = 1024; p.K = 512; p.zdiv = 1;
    p.ksp = kbuf; p.vT = vTb; p.H = H; p.dh = dh;
    dim3 g(16, M / 128, 1);
    gemm_k<1><<<g, 256, 0, stream>>>(p);
  };
  auto ln = [&](const float* src, long srcB, int rowsPerB, int rows,
                const float* s, const float* b, const float* g, const float* be,
                float* dst) {
    ln_k<<<rows, 256, 0, stream>>>(src, srcB, rowsPerB, s, b, g, be, dst);
  };

  auto attention = [&](const float* qsrc, const float* kvsrc,
                       const bf16_t* wqT, const bf16_t* wkvT, const bf16_t* woT,
                       const float* bo, float* xio) {
    gemm(qsrc, 512, 0, 0, wqT, 512, 0, 0, qbuf, 512, 0, 0,
         8192, 512, 512, 1, 1, nullptr, nullptr, nullptr, nullptr, nullptr);
    gemmKV(kvsrc, wkvT, 8192, NHEAD, DHEAD);
    // S[b,h] = Q[b,:,h,:] * K[b,:,h,:]^T  (z = b*8+h)
    gemm(qbuf, 512, 262144, 64, kbuf, 512, 262144, 64,
         Sbuf, 512, 2097152, 262144, 512, 512, 64, 128, 8,
         nullptr, nullptr, nullptr, nullptr, nullptr);
    softmax_k<<<65536, 256, 0, stream>>>(Sbuf, 0.125f);  // 64^-0.5
    // O[b,:,h,:] = P[b,h] * V[b,h]   (vT: [b][h][d][j])
    gemm(Sbuf, 512, 2097152, 262144, vTb, 512, 262144, 32768,
         obuf, 512, 262144, 64, 512, 64, 512, 128, 8,
         nullptr, nullptr, nullptr, nullptr, nullptr);
    gemm(obuf, 512, 0, 0, woT, 512, 0, 0, xio, 512, 0, 0,
         8192, 512, 512, 1, 1, bo, xio, nullptr, nullptr, nullptr);
  };
  auto ffn = [&](const float* xnrm, const bf16_t* w1T, const float* b1,
                 const bf16_t* w2T, const float* b2, float* xio) {
    gemm(xnrm, 512, 0, 0, w1T, 512, 0, 0, Sbuf, 4096, 0, 0,
         8192, 4096, 512, 1, 1, b1, nullptr, nullptr, nullptr, nullptr);
    geglu_k<<<65536, 256, 0, stream>>>(Sbuf, obuf);
    gemm(obuf, 2048, 0, 0, w2T, 2048, 0, 0, xio, 512, 0, 0,
         8192, 512, 2048, 1, 1, b2, xio, nullptr, nullptr, nullptr);
  };

  // ---- point-embed context, init latents ----
  embed_k<<<(32768 * 64) / 256, 256, 0, stream>>>(IN[I_CPTS], emb, 32768);
  gemm(emb, 64, 0, 0, wPe, 64, 0, 0, ctx, 512, 0, 0,
       32768, 512, 64, 1, 1, IN[I_PE_B], nullptr,
       IN[I_CVALS], IN[I_VE_W], IN[I_VE_B]);
  bcast_k<<<16384, 256, 0, stream>>>(IN[I_LATENT], xbuf);

  // ---- layers ----
  for (int i = 0; i < NDEPTH; i++) {
    long v = (long)i * 512;
    // cross-attention to context chunk i
    ln(ctx + (long)i * 262144, 1048576, 512, 8192,
       IN[I_CA_LNC_S] + v, IN[I_CA_LNC_B] + v, nullptr, nullptr, cnb);
    ln(xbuf, 262144, 512, 8192, IN[I_CA_LN_S] + v, IN[I_CA_LN_B] + v,
       nullptr, nullptr, xn);
    attention(xn, cnb, wCaWq[i], wCaWkv[i], wCaWo[i], IN[I_CA_BO] + v, xbuf);
    // FF
    ln(xbuf, 262144, 512, 8192, IN[I_CF_LN_S] + v, IN[I_CF_LN_B] + v,
       nullptr, nullptr, xn);
    ffn(xn, wCfW1[i], IN[I_CF_B1] + (long)i * 4096, wCfW2[i], IN[I_CF_B2] + v, xbuf);
    // FiLM self-attention
    ln(xbuf, 262144, 512, 8192, IN[I_SA_LN_S] + v, IN[I_SA_LN_B] + v,
       film + (i * 4 + 0) * 8192L, film + (i * 4 + 1) * 8192L, xn);
    attention(xn, xn, wSaWq[i], wSaWkv[i], wSaWo[i], IN[I_SA_BO] + v, xbuf);
    // FiLM FF
    ln(xbuf, 262144, 512, 8192, IN[I_SF_LN_S] + v, IN[I_SF_LN_B] + v,
       film + (i * 4 + 2) * 8192L, film + (i * 4 + 3) * 8192L, xn);
    ffn(xn, wSfW1[i], IN[I_SF_B1] + (long)i * 4096, wSfW2[i], IN[I_SF_B2] + v, xbuf);
  }

  // ---- decode: single-head cross attention over latents ----
  embed_k<<<(65536 * 64) / 256, 256, 0, stream>>>(IN[I_QUERIES], emb, 65536);
  gemm(emb, 64, 0, 0, wPe, 64, 0, 0, Sbuf, 512, 0, 0,
       65536, 512, 64, 1, 1, IN[I_PE_B], nullptr, nullptr, nullptr, nullptr);
  ln(Sbuf, 2097152, 4096, 65536, IN[I_DEC_LN_S], IN[I_DEC_LN_B],
     nullptr, nullptr, Sbuf);  // qn (in place)
  ln(xbuf, 262144, 512, 8192, IN[I_DEC_LNC_S], IN[I_DEC_LNC_B],
     nullptr, nullptr, cnb);
  gemm(Sbuf, 512, 0, 0, wDecWq, 512, 0, 0, qbuf, 512, 0, 0,
       65536, 512, 512, 1, 1, nullptr, nullptr, nullptr, nullptr, nullptr);
  gemmKV(cnb, wDecWkv, 8192, 1, 512);  // k bf16 [b][512][512], vT [b][512][512]
  gemm(qbuf, 512, 2097152, 0, kbuf, 512, 262144, 0,
       Sbuf, 512, 2097152, 0, 4096, 512, 512, 16, 1,
       nullptr, nullptr, nullptr, nullptr, nullptr);
  softmax_k<<<65536, 256, 0, stream>>>(Sbuf, 0.04419417382415922f);  // 512^-0.5
  gemm(Sbuf, 512, 2097152, 0, vTb, 512, 262144, 0,
       obuf, 512, 2097152, 0, 4096, 512, 512, 16, 1,
       nullptr, nullptr, nullptr, nullptr, nullptr);
  gemm(obuf, 512, 0, 0, wDecWo, 512, 0, 0, qbuf, 512, 0, 0,
       65536, 512, 512, 1, 1, IN[I_DEC_BO], nullptr, nullptr, nullptr, nullptr);
  ln(qbuf, 2097152, 4096, 65536, IN[I_OUT_LN_S], IN[I_OUT_LN_B],
     film + 16L * 8192, film + 17L * 8192, qbuf);
  gemv_out_k<<<65536, 128, 0, stream>>>(qbuf, IN[I_OUT_W], IN[I_OUT_B],
                                        (float*)d_out);
}
